// AAConvLayer_75582834475557
// MI455X (gfx1250) — compile-verified, matrix operand order checked by reading, not run on hardware
//
#include <hip/hip_runtime.h>
#include <math.h>

typedef __attribute__((ext_vector_type(16))) __bf16 v16bf;
typedef __attribute__((ext_vector_type(8)))  float  v8f;

#define B_     4
#define CIN    128
#define COUT   128
#define STYLE_ 512
#define H_     64
#define W_     64
#define N_     (H_*W_)

__device__ __forceinline__ unsigned short f2bf(float f) {
  unsigned int u = __float_as_uint(f);
  u += 0x7FFFu + ((u >> 16) & 1u);
  return (unsigned short)(u >> 16);
}

__device__ __forceinline__ v16bf ld_frag(const unsigned short* p) {
  return *(const v16bf*)p;
}

// ---- gfx1250 async global->LDS copy (16B per lane), tracked by ASYNCcnt
__device__ __forceinline__ void async_copy16(void* lds, const void* g) {
  unsigned lds_off = (unsigned)(size_t)lds;   // LDS aperture: low 32 bits = LDS offset
  asm volatile("global_load_async_to_lds_b128 %0, %1, off"
               :: "v"(lds_off), "v"(g) : "memory");
}
__device__ __forceinline__ void async_wait() {
  asm volatile("s_wait_asynccnt 0x0" ::: "memory");
}

// ---- style modulation: scale = 1 + style@ss_w^T + ss_b ; bias = style@sb_w^T + sb_b
__global__ void kstyle(const float* style, const float* ss_w, const float* ss_b,
                       const float* sb_w, const float* sb_b,
                       float* sbuf, float* bbuf) {
  int b = blockIdx.x, c = threadIdx.x;
  const float4* st = (const float4*)(style + b*STYLE_);
  const float4* sw = (const float4*)(ss_w + c*STYLE_);
  const float4* bw = (const float4*)(sb_w + c*STYLE_);
  float s = 0.f, t = 0.f;
  for (int i = 0; i < STYLE_/4; ++i) {
    float4 sv = st[i], a = sw[i], d = bw[i];
    s += sv.x*a.x + sv.y*a.y + sv.z*a.z + sv.w*a.w;
    t += sv.x*d.x + sv.y*d.y + sv.z*d.z + sv.w*d.w;
  }
  sbuf[b*COUT + c] = 1.0f + s + ss_b[c];
  bbuf[b*COUT + c] = t + sb_b[c];
}

// ---- conv weights f32 [co][ci][3][3] -> bf16 [kk][co][ci]
__global__ void kconvw(const float* w, unsigned short* wb) {
  int idx = blockIdx.x*256 + threadIdx.x;
  if (idx >= COUT*CIN*9) return;
  int co = idx / (CIN*9);
  int rem = idx % (CIN*9);
  int ci = rem / 9;
  int kk = rem % 9;
  wb[(kk*COUT + co)*CIN + ci] = f2bf(w[idx]);
}

// ---- fused qkv weights -> bf16 [160][128] (rows: 0..15 q, 16..31 k, 32..159 v)
__global__ void kqkvw(const float* qw, const float* kw, const float* vw,
                      unsigned short* wqkv) {
  int idx = blockIdx.x*256 + threadIdx.x;
  if (idx >= 160*CIN) return;
  int o = idx / CIN, ci = idx % CIN;
  float v;
  if (o < 16)      v = qw[o*CIN + ci];
  else if (o < 32) v = kw[(o-16)*CIN + ci];
  else             v = vw[(o-32)*CIN + ci];
  wqkv[idx] = f2bf(v);
}

// ---- x -> zero-padded channels-last bf16 [b][66][66][128]
__global__ void kxpad(const float* x, unsigned short* xpad) {
  int idx = blockIdx.x*256 + threadIdx.x;
  const int total = B_*66*66*CIN;
  if (idx >= total) return;
  int ci = idx % CIN;
  int r = idx / CIN;
  int xx = r % 66; r /= 66;
  int yy = r % 66; int b = r / 66;
  float v = 0.f;
  if (yy >= 1 && yy <= H_ && xx >= 1 && xx <= W_)
    v = x[((b*CIN + ci)*H_ + (yy-1))*W_ + (xx-1)];
  xpad[idx] = f2bf(v);
}

// ---- conv3x3 (implicit GEMM, WMMA bf16) + conv bias + AdaIN modulation
__global__ __launch_bounds__(256) void kconv(
    const unsigned short* xpad, const unsigned short* wb,
    const float* conv_b, const float* sbuf, const float* bbuf,
    float* hmod, unsigned short* hbf) {
  __shared__ __align__(32) unsigned short xs[3*66*CIN];   // 3 padded input rows
  __shared__ __align__(32) unsigned short wt[COUT*32];    // one (kk, 32-ci) weight slab
  const int y = blockIdx.x, b = blockIdx.y;
  const int tid = threadIdx.x;
  const int lane = tid & 31, wv = tid >> 5;
  const int li = lane & 15, half = lane >> 4;

  { // rows y..y+2 of xpad are contiguous: 3168 x 16B, async to LDS
    const uint4* s4 = (const uint4*)(xpad + ((size_t)(b*66 + y)*66)*CIN);
    uint4* d4 = (uint4*)xs;
#pragma unroll
    for (int i = 0; i < 12; ++i) async_copy16(&d4[tid + i*256], &s4[tid + i*256]);
    if (tid < 96) async_copy16(&d4[3072 + tid], &s4[3072 + tid]);
    async_wait();
  }

  v8f acc[4];
#pragma unroll
  for (int r = 0; r < 4; ++r) acc[r] = v8f{0,0,0,0,0,0,0,0};

  const uint4* wsrc4 = (const uint4*)wb;  // 16 uint4 per [kk][co] row
  for (int it = 0; it < 36; ++it) {
    const int kk = it >> 2, cb = it & 3;
    const int ky = kk / 3, kx = kk % 3;
    __syncthreads();
    { // stage wt[co][0..31] = wb[kk][co][cb*32..+32] : 512 x 16B async
      uint4* d4 = (uint4*)wt;
#pragma unroll
      for (int u = 0; u < 2; ++u) {
        int f = tid + u*256;
        int co = f >> 2, i = f & 3;
        async_copy16(&d4[f], &wsrc4[(kk*COUT + co)*16 + cb*4 + i]);
      }
      if (it + 1 < 36) {   // prefetch next slab into L2
        int it2 = it + 1;
        int kk2 = it2 >> 2, cb2 = it2 & 3;
        int co = tid >> 2, i = tid & 3;
        __builtin_prefetch((const void*)&wsrc4[(kk2*COUT + co)*16 + cb2*4 + i], 0, 0);
      }
      async_wait();
    }
    __syncthreads();
    v16bf bfr = ld_frag(wt + (wv*16 + li)*32 + half*16);
#pragma unroll
    for (int r = 0; r < 4; ++r) {
      int xv = r*16 + li + kx;               // 0..65
      v16bf af = ld_frag(xs + (ky*66 + xv)*CIN + cb*32 + half*16);
      acc[r] = __builtin_amdgcn_wmma_f32_16x16x32_bf16(
          false, af, false, bfr, (short)0, acc[r], false, false);
    }
  }

  const int co = wv*16 + li;
  const float cbv = conv_b[co];
  const float ms  = sbuf[b*COUT + co];
  const float bb  = bbuf[b*COUT + co];
  float4* hp = (float4*)(hmod + ((size_t)(b*COUT + co))*N_ + y*W_);
#pragma unroll
  for (int r = 0; r < 4; ++r) {
#pragma unroll
    for (int g = 0; g < 2; ++g) {
      float4 v4;
      float* vp = (float*)&v4;
#pragma unroll
      for (int j = 0; j < 4; ++j) {
        float v = (acc[r][g*4 + j] + cbv) * ms + bb;
        vp[j] = v;
        int n = y*W_ + r*16 + half*8 + g*4 + j;
        hbf[((size_t)(b*N_ + n))*CIN + co] = f2bf(v);     // bf16 [b][n][c]
      }
      hp[r*4 + half*2 + g] = v4;                          // f32 [b][c][n]
    }
  }
}

// ---- fused q/k/v 1x1 projection (GEMM 64x160x128, WMMA bf16)
__global__ __launch_bounds__(256) void kqkv(
    const unsigned short* hbf, const unsigned short* wqkv,
    const float* qb, const float* kb, const float* vb,
    unsigned short* qbuf, unsigned short* kbuf, unsigned short* vbuf) {
  __shared__ __align__(32) unsigned short Ah[64*CIN];     // 1024 x 16B
  __shared__ __align__(32) unsigned short Bw[160*CIN];    // 2560 x 16B
  const int t = blockIdx.x, b = blockIdx.y;
  const int nb = t*64;
  const int tid = threadIdx.x, lane = tid & 31, wv = tid >> 5;
  const int li = lane & 15, half = lane >> 4;
  {
    const uint4* s4 = (const uint4*)(hbf + ((size_t)(b*N_ + nb))*CIN);
    uint4* d4 = (uint4*)Ah;
#pragma unroll
    for (int i = 0; i < 4; ++i) async_copy16(&d4[tid + i*256], &s4[tid + i*256]);
    const uint4* w4 = (const uint4*)wqkv;
    uint4* e4 = (uint4*)Bw;
#pragma unroll
    for (int i = 0; i < 10; ++i) async_copy16(&e4[tid + i*256], &w4[tid + i*256]);
    async_wait();
  }
  __syncthreads();
  const int ncb = (wv < 2) ? 2 : 1;
  for (int q = 0; q < ncb; ++q) {
    int cb = (q == 0) ? wv : (8 + wv);
    v8f acc[4];
#pragma unroll
    for (int r = 0; r < 4; ++r) acc[r] = v8f{0,0,0,0,0,0,0,0};
#pragma unroll
    for (int kc = 0; kc < 4; ++kc) {
      v16bf bfr = ld_frag(Bw + (cb*16 + li)*CIN + kc*32 + half*16);
#pragma unroll
      for (int r = 0; r < 4; ++r) {
        v16bf af = ld_frag(Ah + (r*16 + li)*CIN + kc*32 + half*16);
        acc[r] = __builtin_amdgcn_wmma_f32_16x16x32_bf16(
            false, af, false, bfr, (short)0, acc[r], false, false);
      }
    }
    int co = cb*16 + li;
    if (co < 32) {                       // q / k, strided b16 stores (small)
      float bias = (co < 16) ? qb[co] : kb[co-16];
      unsigned short* dst = (co < 16) ? qbuf : kbuf;
      int cc = co & 15;
#pragma unroll
      for (int r = 0; r < 4; ++r)
#pragma unroll
        for (int j = 0; j < 8; ++j) {
          int n = nb + r*16 + half*8 + j;
          dst[((size_t)(b*N_ + n))*32 + cc] = f2bf(acc[r][j] + bias);
        }
    } else {                             // v: [b][c][n], 8 contiguous bf16 -> uint4
      float bias = vb[co-32];
      unsigned short* dst = vbuf + ((size_t)(b*COUT + (co-32)))*N_ + nb;
#pragma unroll
      for (int r = 0; r < 4; ++r) {
        uint4 pk;
        unsigned int* pu = (unsigned int*)&pk;
#pragma unroll
        for (int d = 0; d < 4; ++d) {
          unsigned int lo = f2bf(acc[r][d*2]   + bias);
          unsigned int hi = f2bf(acc[r][d*2+1] + bias);
          pu[d] = lo | (hi << 16);
        }
        *(uint4*)(dst + r*16 + half*8) = pk;
      }
    }
  }
}

// ---- flash attention (64q x 64k tiles) + gamma*out + h (in place on hmod)
__global__ __launch_bounds__(256) void kattn(
    const unsigned short* qbuf, const unsigned short* kbuf,
    const unsigned short* vbuf, const float* gamma, float* hmod) {
  __shared__ __align__(32) unsigned short Qs[64*32];
  __shared__ __align__(32) unsigned short Ks[64*32];
  __shared__ __align__(32) unsigned short Vt[COUT*64];
  __shared__ __align__(32) float Sf[64*66];          // stride 66: conflict-free
  __shared__ __align__(32) unsigned short Pb[64*80]; // stride 80: 32B-aligned rows
  __shared__ float rowA[64], rowMsh[64], rowL[64];
  __shared__ float pmax[256], psum[256];

  const int qt = blockIdx.x, b = blockIdx.y;
  const int nqb = qt*64;
  const int tid = threadIdx.x, lane = tid & 31, wv = tid >> 5;
  const int li = lane & 15, half = lane >> 4;

  {
    const uint4* s4 = (const uint4*)(qbuf + ((size_t)(b*N_ + nqb))*32);
    async_copy16(&((uint4*)Qs)[tid], &s4[tid]);      // 256 x 16B exactly
  }

  float mReg = -INFINITY, lReg = 0.f, aReg = 0.f;
  v8f o[4];
#pragma unroll
  for (int r = 0; r < 4; ++r) o[r] = v8f{0,0,0,0,0,0,0,0};

  const int sr = wv & 3, scs = wv >> 2;
  const int srow = tid >> 2, sseg = tid & 3;

  for (int kt = 0; kt < 64; ++kt) {
    const int mb = kt*64;
    __syncthreads();
    const unsigned short* ksrc = kbuf + ((size_t)(b*N_ + mb))*32;
    const unsigned short* vsrc = vbuf + ((size_t)(b*COUT + (tid >> 1)))*N_ + mb;
    {
      async_copy16(&((uint4*)Ks)[tid], &((const uint4*)ksrc)[tid]);
      int c = tid >> 1, hh = tid & 1;
      const uint4* vs4 = (const uint4*)vsrc;
      uint4* vd4 = (uint4*)(Vt + c*64);
#pragma unroll
      for (int i = 0; i < 4; ++i) async_copy16(&vd4[hh*4 + i], &vs4[hh*4 + i]);
      if (kt < 63) {                      // prefetch next K/V tiles into L2
        __builtin_prefetch((const void*)(ksrc + 2048 + tid*8), 0, 0);
        __builtin_prefetch((const void*)(vsrc + 64), 0, 0);
      }
      async_wait();
    }
    __syncthreads();
    // S = Q K^T : 16 tiles, 2 per wave
#pragma unroll
    for (int i2 = 0; i2 < 2; ++i2) {
      int c = scs*2 + i2;
      v16bf af = ld_frag(Qs + (sr*16 + li)*32 + half*16);
      v16bf bfr = ld_frag(Ks + (c*16 + li)*32 + half*16);
      v8f z = v8f{0,0,0,0,0,0,0,0};
      v8f s = __builtin_amdgcn_wmma_f32_16x16x32_bf16(
          false, af, false, bfr, (short)0, z, false, false);
#pragma unroll
      for (int j = 0; j < 8; ++j)
        Sf[(sr*16 + half*8 + j)*66 + c*16 + li] = s[j];
    }
    __syncthreads();
    { // pass A: segment maxes, all 256 threads
      float m = -INFINITY;
#pragma unroll
      for (int i = 0; i < 16; ++i) m = fmaxf(m, Sf[srow*66 + sseg*16 + i]);
      pmax[tid] = m;
    }
    __syncthreads();
    if (tid < 64) { // pass B: row max + alpha (register row state)
      float mx = fmaxf(fmaxf(pmax[tid*4], pmax[tid*4+1]),
                       fmaxf(pmax[tid*4+2], pmax[tid*4+3]));
      mx = fmaxf(mx, mReg);
      aReg = __expf(mReg - mx);
      mReg = mx;
      rowA[tid] = aReg;
      rowMsh[tid] = mx;
    }
    __syncthreads();
    { // pass C: exp + P(bf16) + segment sums, all 256 threads
      float mm = rowMsh[srow];
      float sum = 0.f;
#pragma unroll
      for (int i = 0; i < 16; ++i) {
        float p = __expf(Sf[srow*66 + sseg*16 + i] - mm);
        sum += p;
        Pb[srow*80 + sseg*16 + i] = f2bf(p);
      }
      psum[tid] = sum;
    }
    __syncthreads();
    if (tid < 64)   // pass D: running denominator (registers only)
      lReg = lReg*aReg + (psum[tid*4] + psum[tid*4+1] + psum[tid*4+2] + psum[tid*4+3]);
    // O = O*alpha + P @ V^T (wave wv owns channel block wv*16)
#pragma unroll
    for (int r = 0; r < 4; ++r) {
#pragma unroll
      for (int j = 0; j < 8; ++j) o[r][j] *= rowA[r*16 + half*8 + j];
#pragma unroll
      for (int kc = 0; kc < 2; ++kc) {
        v16bf af = ld_frag(Pb + (r*16 + li)*80 + kc*32 + half*16);
        v16bf bfr = ld_frag(Vt + (wv*16 + li)*64 + kc*32 + half*16);
        o[r] = __builtin_amdgcn_wmma_f32_16x16x32_bf16(
            false, af, false, bfr, (short)0, o[r], false, false);
      }
    }
  }
  if (tid < 64) rowL[tid] = lReg;
  __syncthreads();
  const float g = gamma[0];
  const int c = wv*16 + li;
  float4* hp = (float4*)(hmod + ((size_t)(b*COUT + c))*N_ + nqb);
#pragma unroll
  for (int r = 0; r < 4; ++r)
#pragma unroll
    for (int gq = 0; gq < 2; ++gq) {
      float4 old = hp[r*4 + half*2 + gq];
      float* op = (float*)&old;
#pragma unroll
      for (int j = 0; j < 4; ++j) {
        int row = r*16 + half*8 + gq*4 + j;
        op[j] = g * (o[r][gq*4 + j] / rowL[row]) + op[j];
      }
      hp[r*4 + half*2 + gq] = old;
    }
}

// ---- InstanceNorm (biased var, eps=1e-5) + LeakyReLU(0.2)
__global__ __launch_bounds__(256) void knorm(const float* hf, float* out) {
  __shared__ float rs[256], rs2[256];
  int bc = blockIdx.x;
  const float4* p4 = (const float4*)(hf + (size_t)bc*N_);
  float s = 0.f, s2 = 0.f;
  int tid = threadIdx.x;
#pragma unroll
  for (int i = 0; i < 4; ++i) {
    float4 v = p4[tid + i*256];
    s  += v.x + v.y + v.z + v.w;
    s2 += v.x*v.x + v.y*v.y + v.z*v.z + v.w*v.w;
  }
  rs[tid] = s; rs2[tid] = s2;
  __syncthreads();
  for (int off = 128; off > 0; off >>= 1) {
    if (tid < off) { rs[tid] += rs[tid+off]; rs2[tid] += rs2[tid+off]; }
    __syncthreads();
  }
  float mu  = rs[0] * (1.0f/N_);
  float var = rs2[0] * (1.0f/N_) - mu*mu;
  float inv = rsqrtf(var + 1e-5f);
  float4* q4 = (float4*)(out + (size_t)bc*N_);
#pragma unroll
  for (int i = 0; i < 4; ++i) {
    float4 v = p4[tid + i*256];
    float* vp = (float*)&v;
#pragma unroll
    for (int j = 0; j < 4; ++j) {
      float y = (vp[j] - mu) * inv;
      vp[j] = (y >= 0.f) ? y : 0.2f*y;
    }
    q4[tid + i*256] = v;
  }
}

extern "C" void kernel_launch(void* const* d_in, const int* in_sizes, int n_in,
                              void* d_out, int out_size, void* d_ws, size_t ws_size,
                              hipStream_t stream) {
  (void)in_sizes; (void)n_in; (void)out_size; (void)ws_size;
  const float* x      = (const float*)d_in[0];
  const float* style  = (const float*)d_in[1];
  const float* conv_w = (const float*)d_in[2];
  const float* conv_b = (const float*)d_in[3];
  const float* ss_w   = (const float*)d_in[4];
  const float* ss_b   = (const float*)d_in[5];
  const float* sb_w   = (const float*)d_in[6];
  const float* sb_b   = (const float*)d_in[7];
  const float* q_w    = (const float*)d_in[8];
  const float* q_b    = (const float*)d_in[9];
  const float* k_w    = (const float*)d_in[10];
  const float* k_b    = (const float*)d_in[11];
  const float* v_w    = (const float*)d_in[12];
  const float* v_b    = (const float*)d_in[13];
  const float* gamma  = (const float*)d_in[14];
  float* out = (float*)d_out;

  char* ws = (char*)d_ws;
  float*          hmod = (float*)(ws + 0);                    //  8.0 MB f32 [b][c][n]
  unsigned short* hbf  = (unsigned short*)(ws + 8388608);     //  4.0 MB bf16 [b][n][c]
  unsigned short* qbuf = (unsigned short*)(ws + 12582912);    //  1.0 MB bf16 [b][n][32]
  unsigned short* kbuf = (unsigned short*)(ws + 13631488);    //  1.0 MB bf16 [b][n][32]
  unsigned short* vbuf = (unsigned short*)(ws + 14680064);    //  4.0 MB bf16 [b][c][n]
  unsigned short* xpad = (unsigned short*)(ws + 18874368);    //  4.3 MB bf16 padded x
  unsigned short* wb   = (unsigned short*)(ws + 23334912);    //  288 KB
  unsigned short* wqkv = (unsigned short*)(ws + 23629824);    //   40 KB
  float*          sbuf = (float*)(ws + 23670784);
  float*          bbuf = (float*)(ws + 23672832);

  hipMemsetAsync(qbuf, 0, 2*1048576, stream);  // zero K-padding of q/k (cols 16..31)
  kstyle<<<dim3(B_), dim3(COUT), 0, stream>>>(style, ss_w, ss_b, sb_w, sb_b, sbuf, bbuf);
  kconvw<<<dim3((COUT*CIN*9 + 255)/256), dim3(256), 0, stream>>>(conv_w, wb);
  kqkvw<<<dim3((160*CIN + 255)/256), dim3(256), 0, stream>>>(q_w, k_w, v_w, wqkv);
  kxpad<<<dim3((B_*66*66*CIN + 255)/256), dim3(256), 0, stream>>>(x, xpad);
  kconv<<<dim3(H_, B_), dim3(256), 0, stream>>>(xpad, wb, conv_b, sbuf, bbuf, hmod, hbf);
  kqkv<<<dim3(N_/64, B_), dim3(256), 0, stream>>>(hbf, wqkv, q_b, k_b, v_b, qbuf, kbuf, vbuf);
  kattn<<<dim3(N_/64, B_), dim3(256), 0, stream>>>(qbuf, kbuf, vbuf, gamma, hmod);
  knorm<<<dim3(B_*COUT), dim3(256), 0, stream>>>(hmod, out);
}
